// MultiHeadAttention_11733850652978
// MI455X (gfx1250) — compile-verified, matrix operand order checked by reading, not run on hardware
//
#include <hip/hip_runtime.h>
#include <hip/hip_bf16.h>

// ---------------------------------------------------------------------------
// MHA block for MI455X (gfx1250, wave32, WMMA).
// B=4, T=2048, C=1024, H=16, HD=64.  All matmuls on v_wmma_f32_16x16x32_bf16.
// GEMMs: 128x128 block tiles, K=64 slabs double-buffered in LDS via
// global_load_async_to_lds_b128 (ASYNCcnt); 144B LDS row pitch (conflict-free).
// Attention: flash-style, 32 query rows per wave, 32 keys per step.
// ---------------------------------------------------------------------------

typedef __attribute__((ext_vector_type(16))) __bf16 v16bf;
typedef __attribute__((ext_vector_type(8)))  float  v8f;

#define B_  4
#define T_  2048
#define C_  1024
#define H_  16
#define HD_ 64
#define M_  (B_ * T_)       // 8192
#define N1_ (3 * C_)        // 3072

#define PITCH 72            // bf16 elems per LDS tile row (144 B): r*36 mod 64 full orbit

union FragU { v16bf v; uint4 u[2]; };

__device__ __forceinline__ v8f wmma_bf16(v16bf a, v16bf b, v8f c) {
  return __builtin_amdgcn_wmma_f32_16x16x32_bf16(false, a, false, b,
                                                 (short)0, c, false, false);
}

__device__ __forceinline__ v8f zero8() { v8f z = {0.f,0.f,0.f,0.f,0.f,0.f,0.f,0.f}; return z; }

// 16x32 bf16 fragment, ISA "16-bit A-matrix 16x32" layout, from global memory.
__device__ __forceinline__ v16bf load_frag(const __hip_bfloat16* __restrict__ base,
                                           int ld, int rowbase, int kbase) {
  const int lane  = threadIdx.x & 31;
  const int r     = lane & 15;
  const int half  = lane >> 4;
  const __hip_bfloat16* p = base + (size_t)(rowbase + r) * ld + kbase + half * 8;
  FragU f;
  f.u[0] = *(const uint4*)(p);
  f.u[1] = *(const uint4*)(p + 16);
  return f.v;
}

// Same fragment out of an LDS tile (PITCH row pitch, K=64 tile, kbase 0 or 32).
__device__ __forceinline__ v16bf lds_frag(const __hip_bfloat16* tile,
                                          int rowbase, int kbase) {
  const int lane  = threadIdx.x & 31;
  const int r     = lane & 15;
  const int half  = lane >> 4;
  const __hip_bfloat16* p = tile + (rowbase + r) * PITCH + kbase + half * 8;
  FragU f;
  f.u[0] = *(const uint4*)(p);
  f.u[1] = *(const uint4*)(p + 16);
  return f.v;
}

// ---- CDNA5 async global->LDS copy (ASYNCcnt) ------------------------------
__device__ __forceinline__ void async_copy16(void* lds_dst, const void* gsrc) {
  unsigned lds_off = (unsigned)(uintptr_t)lds_dst;
  asm volatile("global_load_async_to_lds_b128 %0, %1, off"
               :: "v"(lds_off), "v"(gsrc)
               : "memory");
}
__device__ __forceinline__ void wait_asynccnt0() {
  asm volatile("s_wait_asynccnt 0x0" ::: "memory");
}

// Stage a 128x64 bf16 tile (rows row0.., cols kk..kk+63, leading dim ld) into
// an LDS tile with PITCH pitch.  1024 x 16B units, 256 threads -> 4 per thread.
__device__ __forceinline__ void stage_tile64(const __hip_bfloat16* __restrict__ g,
                                             int ld, int row0, int kk,
                                             __hip_bfloat16* tile, int tid) {
#pragma unroll
  for (int rep = 0; rep < 4; ++rep) {
    const int u   = tid + rep * 256;
    const int row = u >> 3;
    const int seg = u & 7;
    async_copy16(tile + row * PITCH + seg * 8,
                 g + (size_t)(row0 + row) * ld + kk + seg * 8);
  }
}

// ---------------------------------------------------------------------------
// Stage 0: fp32 -> bf16 converts (weights transposed to [N][K]).
// ---------------------------------------------------------------------------
__global__ void cvt_x_kernel(const float* __restrict__ x,
                             __hip_bfloat16* __restrict__ xb, int n) {
  for (int i = blockIdx.x * blockDim.x + threadIdx.x; i < n;
       i += gridDim.x * blockDim.x)
    xb[i] = __float2bfloat16(x[i]);
}

__global__ void cvt_T_kernel(const float* __restrict__ w,          // [K][N]
                             __hip_bfloat16* __restrict__ wT,      // [N][K]
                             int K, int N) {
  int total = K * N;
  for (int i = blockIdx.x * blockDim.x + threadIdx.x; i < total;
       i += gridDim.x * blockDim.x) {
    int n = i / K;
    int k = i - n * K;
    wT[i] = __float2bfloat16(w[(size_t)k * N + n]);
  }
}

// ---------------------------------------------------------------------------
// Stage 1: QKV GEMM [8192x1024]x[1024x3072]+bias.  Block 128x128, 8 waves,
// wave tile 32x64, K=64 double-buffered slabs (16 WMMA per barrier).
// Epilogue scatters to Q,K [B,H,T,HD] and transposed V [B,H,HD,T].
// ---------------------------------------------------------------------------
__global__ void __launch_bounds__(256)
gemm_qkv_kernel(const __hip_bfloat16* __restrict__ xb,   // [M][C]
                const __hip_bfloat16* __restrict__ wT,   // [3C][C]
                const float* __restrict__ bias,          // [3C]
                __hip_bfloat16* __restrict__ q,
                __hip_bfloat16* __restrict__ k,
                __hip_bfloat16* __restrict__ vT) {
  __shared__ alignas(16) __hip_bfloat16 ldsA[2][128 * PITCH];
  __shared__ alignas(16) __hip_bfloat16 ldsB[2][128 * PITCH];

  const int tid    = threadIdx.x;
  const int lane   = tid & 31;
  const int wave   = tid >> 5;
  const int lane15 = lane & 15;
  const int half   = lane >> 4;
  const int wm = wave & 3;            // 4 waves along M
  const int wn = wave >> 2;           // 2 waves along N
  const int mblk = blockIdx.y * 128;
  const int nblk = blockIdx.x * 128;

  v8f acc[2][4];
#pragma unroll
  for (int mi = 0; mi < 2; ++mi)
#pragma unroll
    for (int ni = 0; ni < 4; ++ni) acc[mi][ni] = zero8();

  stage_tile64(xb, C_, mblk, 0, &ldsA[0][0], tid);
  stage_tile64(wT, C_, nblk, 0, &ldsB[0][0], tid);
  wait_asynccnt0();
  __syncthreads();

  int bi = 0;
  for (int kk = 0; kk < C_; kk += 64, bi ^= 1) {
    if (kk + 64 < C_) {
      stage_tile64(xb, C_, mblk, kk + 64, &ldsA[bi ^ 1][0], tid);
      stage_tile64(wT, C_, nblk, kk + 64, &ldsB[bi ^ 1][0], tid);
    }
#pragma unroll
    for (int ks = 0; ks < 64; ks += 32) {
      v16bf a0 = lds_frag(&ldsA[bi][0], wm * 32,      ks);
      v16bf a1 = lds_frag(&ldsA[bi][0], wm * 32 + 16, ks);
      v16bf b0 = lds_frag(&ldsB[bi][0], wn * 64,      ks);
      v16bf b1 = lds_frag(&ldsB[bi][0], wn * 64 + 16, ks);
      v16bf b2 = lds_frag(&ldsB[bi][0], wn * 64 + 32, ks);
      v16bf b3 = lds_frag(&ldsB[bi][0], wn * 64 + 48, ks);
      acc[0][0] = wmma_bf16(a0, b0, acc[0][0]);
      acc[0][1] = wmma_bf16(a0, b1, acc[0][1]);
      acc[0][2] = wmma_bf16(a0, b2, acc[0][2]);
      acc[0][3] = wmma_bf16(a0, b3, acc[0][3]);
      acc[1][0] = wmma_bf16(a1, b0, acc[1][0]);
      acc[1][1] = wmma_bf16(a1, b1, acc[1][1]);
      acc[1][2] = wmma_bf16(a1, b2, acc[1][2]);
      acc[1][3] = wmma_bf16(a1, b3, acc[1][3]);
    }
    wait_asynccnt0();     // my async slab for the next buffer has landed
    __syncthreads();      // everyone done reading bi / writing bi^1
  }

  const int m0 = mblk + wm * 32;
  const int n0 = nblk + wn * 64;
#pragma unroll
  for (int mi = 0; mi < 2; ++mi) {
#pragma unroll
    for (int ni = 0; ni < 4; ++ni) {
      const int ncol = n0 + ni * 16 + lane15;
      const float bv = bias[ncol];
      const int seg = ncol >> 10;       // 0=q 1=k 2=v
      const int c   = ncol & (C_ - 1);
      const int h   = c >> 6;
      const int d   = c & 63;
#pragma unroll
      for (int j = 0; j < 8; ++j) {
        const int m = m0 + mi * 16 + half * 8 + j;
        const int b = m >> 11;
        const int t = m & (T_ - 1);
        const int bh = b * H_ + h;
        const __hip_bfloat16 o = __float2bfloat16(acc[mi][ni][j] + bv);
        if (seg == 0)      q [((size_t)bh * T_ + t) * HD_ + d] = o;
        else if (seg == 1) k [((size_t)bh * T_ + t) * HD_ + d] = o;
        else               vT[((size_t)bh * HD_ + d) * T_ + t] = o;
      }
    }
  }
}

// ---------------------------------------------------------------------------
// Stage 2: causal flash attention per (b,h).  Wave = 32 query rows (two
// 16-row tiles sharing every K/V fragment), 32 keys per step: 8 WMMA for S,
// online softmax, P through per-wave LDS (C->A relayout), 8 WMMA for P*V.
// ---------------------------------------------------------------------------
__global__ void __launch_bounds__(256)
attn_kernel(const __hip_bfloat16* __restrict__ q,    // [B,H,T,HD]
            const __hip_bfloat16* __restrict__ k,    // [B,H,T,HD]
            const __hip_bfloat16* __restrict__ vT,   // [B,H,HD,T]
            __hip_bfloat16* __restrict__ y) {        // [B,T,C]
  __shared__ alignas(16) __hip_bfloat16 plds[8][32 * 32];  // 32 rows x 32 keys per wave

  const int lane   = threadIdx.x & 31;
  const int wave   = threadIdx.x >> 5;
  const int lane15 = lane & 15;
  const int half   = lane >> 4;

  const int qblk = blockIdx.x & 7;      // T/256 = 8 query blocks per (b,h)
  const int bh   = blockIdx.x >> 3;
  const int b    = bh >> 4;
  const int h    = bh & 15;
  const int qbase = qblk * 256 + wave * 32;

  const __hip_bfloat16* qp = q  + (size_t)bh * T_ * HD_;
  const __hip_bfloat16* kp = k  + (size_t)bh * T_ * HD_;
  const __hip_bfloat16* vp = vT + (size_t)bh * HD_ * T_;

  // Resident Q fragments: 2 row-tiles x 2 K-slabs.
  v16bf aq[2][2];
#pragma unroll
  for (int mi = 0; mi < 2; ++mi) {
    aq[mi][0] = load_frag(qp, HD_, qbase + mi * 16, 0);
    aq[mi][1] = load_frag(qp, HD_, qbase + mi * 16, 32);
  }

  float mrow[2][8], lrow[2][8];
#pragma unroll
  for (int mi = 0; mi < 2; ++mi)
#pragma unroll
    for (int j = 0; j < 8; ++j) { mrow[mi][j] = -1e30f; lrow[mi][j] = 0.f; }
  v8f o[2][4];
#pragma unroll
  for (int mi = 0; mi < 2; ++mi)
#pragma unroll
    for (int ni = 0; ni < 4; ++ni) o[mi][ni] = zero8();

  const float scale = 0.125f;                    // 1/sqrt(64)
  const int nsteps = ((qbase + 31) >> 5) + 1;    // causal: keys <= qbase+31

  for (int s = 0; s < nsteps; ++s) {
    const int k0 = s * 32;
    // K fragments loaded once, shared by both row-tiles.
    const v16bf kf00 = load_frag(kp, HD_, k0,      0);
    const v16bf kf01 = load_frag(kp, HD_, k0,      32);
    const v16bf kf10 = load_frag(kp, HD_, k0 + 16, 0);
    const v16bf kf11 = load_frag(kp, HD_, k0 + 16, 32);

    v8f st[2][2];
#pragma unroll
    for (int mi = 0; mi < 2; ++mi) {
      st[mi][0] = zero8();
      st[mi][0] = wmma_bf16(aq[mi][0], kf00, st[mi][0]);
      st[mi][0] = wmma_bf16(aq[mi][1], kf01, st[mi][0]);
      st[mi][1] = zero8();
      st[mi][1] = wmma_bf16(aq[mi][0], kf10, st[mi][1]);
      st[mi][1] = wmma_bf16(aq[mi][1], kf11, st[mi][1]);
    }

#pragma unroll
    for (int mi = 0; mi < 2; ++mi) {
#pragma unroll
      for (int j = 0; j < 8; ++j) {
        const int row = qbase + mi * 16 + half * 8 + j;
        const int c0 = k0 + lane15;
        const int c1 = k0 + 16 + lane15;
        float x0 = (c0 <= row) ? st[mi][0][j] * scale : -1e30f;
        float x1 = (c1 <= row) ? st[mi][1][j] * scale : -1e30f;
        float mx = fmaxf(x0, x1);
        mx = fmaxf(mx, __shfl_xor(mx, 1, 32));
        mx = fmaxf(mx, __shfl_xor(mx, 2, 32));
        mx = fmaxf(mx, __shfl_xor(mx, 4, 32));
        mx = fmaxf(mx, __shfl_xor(mx, 8, 32));
        const float newm = fmaxf(mrow[mi][j], mx);
        const float f  = __expf(mrow[mi][j] - newm);
        const float e0 = __expf(x0 - newm);
        const float e1 = __expf(x1 - newm);
        float rs = e0 + e1;
        rs += __shfl_xor(rs, 1, 32);
        rs += __shfl_xor(rs, 2, 32);
        rs += __shfl_xor(rs, 4, 32);
        rs += __shfl_xor(rs, 8, 32);
        lrow[mi][j] = lrow[mi][j] * f + rs;
        mrow[mi][j] = newm;
#pragma unroll
        for (int ni = 0; ni < 4; ++ni) o[mi][ni][j] *= f;
        // C-layout -> A-layout staging of P
        const int rr = mi * 16 + half * 8 + j;
        plds[wave][rr * 32 + lane15]      = __float2bfloat16(e0);
        plds[wave][rr * 32 + 16 + lane15] = __float2bfloat16(e1);
      }
    }
    asm volatile("s_wait_dscnt 0x0" ::: "memory");
    v16bf pa[2];
#pragma unroll
    for (int mi = 0; mi < 2; ++mi) {
      const __hip_bfloat16* p = &plds[wave][0] + (mi * 16 + lane15) * 32 + half * 8;
      FragU f;
      f.u[0] = *(const uint4*)(p);
      f.u[1] = *(const uint4*)(p + 16);
      pa[mi] = f.v;
    }
    // V fragments loaded once per ni, shared by both row-tiles.
#pragma unroll
    for (int ni = 0; ni < 4; ++ni) {
      const v16bf vf = load_frag(vp, T_, ni * 16, k0);
      o[0][ni] = wmma_bf16(pa[0], vf, o[0][ni]);
      o[1][ni] = wmma_bf16(pa[1], vf, o[1][ni]);
    }
  }

#pragma unroll
  for (int mi = 0; mi < 2; ++mi) {
#pragma unroll
    for (int ni = 0; ni < 4; ++ni) {
      const int c = h * HD_ + ni * 16 + lane15;
#pragma unroll
      for (int j = 0; j < 8; ++j) {
        const int t = qbase + mi * 16 + half * 8 + j;
        y[((size_t)(b * T_ + t)) * C_ + c] =
            __float2bfloat16(o[mi][ni][j] / lrow[mi][j]);
      }
    }
  }
}

// ---------------------------------------------------------------------------
// Stage 3: projection [8192x1024]x[1024x1024]+bias -> fp32 d_out.
// Same K=64 double-buffered scheme.
// ---------------------------------------------------------------------------
__global__ void __launch_bounds__(256)
gemm_proj_kernel(const __hip_bfloat16* __restrict__ yb,  // [M][C]
                 const __hip_bfloat16* __restrict__ wT,  // [C][C]
                 const float* __restrict__ bias,         // [C]
                 float* __restrict__ out) {              // [M][C]
  __shared__ alignas(16) __hip_bfloat16 ldsA[2][128 * PITCH];
  __shared__ alignas(16) __hip_bfloat16 ldsB[2][128 * PITCH];

  const int tid    = threadIdx.x;
  const int lane   = tid & 31;
  const int wave   = tid >> 5;
  const int lane15 = lane & 15;
  const int half   = lane >> 4;
  const int wm = wave & 3;
  const int wn = wave >> 2;
  const int mblk = blockIdx.y * 128;
  const int nblk = blockIdx.x * 128;

  v8f acc[2][4];
#pragma unroll
  for (int mi = 0; mi < 2; ++mi)
#pragma unroll
    for (int ni = 0; ni < 4; ++ni) acc[mi][ni] = zero8();

  stage_tile64(yb, C_, mblk, 0, &ldsA[0][0], tid);
  stage_tile64(wT, C_, nblk, 0, &ldsB[0][0], tid);
  wait_asynccnt0();
  __syncthreads();

  int bi = 0;
  for (int kk = 0; kk < C_; kk += 64, bi ^= 1) {
    if (kk + 64 < C_) {
      stage_tile64(yb, C_, mblk, kk + 64, &ldsA[bi ^ 1][0], tid);
      stage_tile64(wT, C_, nblk, kk + 64, &ldsB[bi ^ 1][0], tid);
    }
#pragma unroll
    for (int ks = 0; ks < 64; ks += 32) {
      v16bf a0 = lds_frag(&ldsA[bi][0], wm * 32,      ks);
      v16bf a1 = lds_frag(&ldsA[bi][0], wm * 32 + 16, ks);
      v16bf b0 = lds_frag(&ldsB[bi][0], wn * 64,      ks);
      v16bf b1 = lds_frag(&ldsB[bi][0], wn * 64 + 16, ks);
      v16bf b2 = lds_frag(&ldsB[bi][0], wn * 64 + 32, ks);
      v16bf b3 = lds_frag(&ldsB[bi][0], wn * 64 + 48, ks);
      acc[0][0] = wmma_bf16(a0, b0, acc[0][0]);
      acc[0][1] = wmma_bf16(a0, b1, acc[0][1]);
      acc[0][2] = wmma_bf16(a0, b2, acc[0][2]);
      acc[0][3] = wmma_bf16(a0, b3, acc[0][3]);
      acc[1][0] = wmma_bf16(a1, b0, acc[1][0]);
      acc[1][1] = wmma_bf16(a1, b1, acc[1][1]);
      acc[1][2] = wmma_bf16(a1, b2, acc[1][2]);
      acc[1][3] = wmma_bf16(a1, b3, acc[1][3]);
    }
    wait_asynccnt0();
    __syncthreads();
  }

  const int m0 = mblk + wm * 32;
  const int n0 = nblk + wn * 64;
#pragma unroll
  for (int mi = 0; mi < 2; ++mi) {
#pragma unroll
    for (int ni = 0; ni < 4; ++ni) {
      const int ncol = n0 + ni * 16 + lane15;
      const float bv = bias[ncol];
#pragma unroll
      for (int j = 0; j < 8; ++j) {
        const int m = m0 + mi * 16 + half * 8 + j;
        out[(size_t)m * C_ + ncol] = acc[mi][ni][j] + bv;
      }
    }
  }
}

// ---------------------------------------------------------------------------
extern "C" void kernel_launch(void* const* d_in, const int* in_sizes, int n_in,
                              void* d_out, int out_size, void* d_ws, size_t ws_size,
                              hipStream_t stream) {
  const float* x      = (const float*)d_in[0];   // [B,T,C]
  const float* w_attn = (const float*)d_in[1];   // [C,3C]
  const float* b_attn = (const float*)d_in[2];   // [3C]
  const float* w_proj = (const float*)d_in[3];   // [C,C]
  const float* b_proj = (const float*)d_in[4];   // [C]
  float* out = (float*)d_out;

  char* ws = (char*)d_ws;
  size_t off = 0;
  __hip_bfloat16* xb     = (__hip_bfloat16*)(ws + off); off += (size_t)M_  * C_  * 2;
  __hip_bfloat16* wattnT = (__hip_bfloat16*)(ws + off); off += (size_t)N1_ * C_  * 2;
  __hip_bfloat16* wprojT = (__hip_bfloat16*)(ws + off); off += (size_t)C_  * C_  * 2;
  __hip_bfloat16* qb     = (__hip_bfloat16*)(ws + off); off += (size_t)M_  * C_  * 2;
  __hip_bfloat16* kb     = (__hip_bfloat16*)(ws + off); off += (size_t)M_  * C_  * 2;
  __hip_bfloat16* vTb    = (__hip_bfloat16*)(ws + off); off += (size_t)M_  * C_  * 2;
  __hip_bfloat16* yb     = (__hip_bfloat16*)(ws + off); off += (size_t)M_  * C_  * 2;
  (void)ws_size; (void)in_sizes; (void)n_in; (void)out_size;

  cvt_x_kernel<<<2048, 256, 0, stream>>>(x, xb, M_ * C_);
  cvt_T_kernel<<<2048, 256, 0, stream>>>(w_attn, wattnT, C_, N1_);
  cvt_T_kernel<<<1024, 256, 0, stream>>>(w_proj, wprojT, C_, C_);

  gemm_qkv_kernel<<<dim3(N1_ / 128, M_ / 128), 256, 0, stream>>>(
      xb, wattnT, b_attn, qb, kb, vTb);

  attn_kernel<<<B_ * H_ * (T_ / 256), 256, 0, stream>>>(qb, kb, vTb, yb);

  gemm_proj_kernel<<<dim3(C_ / 128, M_ / 128), 256, 0, stream>>>(
      yb, wprojT, b_proj, out);
}
